// SoftMoE_52037823758599
// MI455X (gfx1250) — compile-verified, matrix operand order checked by reading, not run on hardware
//
#include <hip/hip_runtime.h>
#include <hip/hip_bf16.h>

// ---------------------------------------------------------------------------
// SoftMoE forward, MI455X (gfx1250).
// GEMMs: f32 (or producer-written bf16) in HBM -> bf16 LDS tiles
//        -> v_wmma_f32_16x16x32_bf16 -> f32 out.
// Block tile 128x128x32, 256 threads (8 wave32), wave tile 64x32 (4x2 WMMA).
// Double-buffered LDS (40KB) with register-staged global loads for overlap.
// Softmax outputs (dispatch/combine) are written directly as bf16: halves
// their HBM traffic and removes A-side cvt work in GEMMs 4 and 8.
// ---------------------------------------------------------------------------

typedef __attribute__((ext_vector_type(16))) __bf16 bf16x16;
typedef __attribute__((ext_vector_type(8)))  __bf16 bf16x8;
typedef __attribute__((ext_vector_type(4)))  __bf16 bf16x4;
typedef __attribute__((ext_vector_type(8)))  float  f32x8;

#define LSTR 40   // LDS row stride in halves: 32 data + 8 pad (80B, 16B-aligned rows)

// Fragment load per CDNA5 16-bit A/B layout:
// lane%16 = row (A) / col (B); halves 0..7 = K = 8*(lane/16)+j ; halves 8..15 = +16.
__device__ __forceinline__ bf16x16 load_frag(const __bf16* s, int rowBase, int lane) {
    const int r  = rowBase + (lane & 15);
    const int kb = (lane >> 4) << 3;
    bf16x8 lo = *reinterpret_cast<const bf16x8*>(s + r * LSTR + kb);        // ds_load_b128
    bf16x8 hi = *reinterpret_cast<const bf16x8*>(s + r * LSTR + 16 + kb);   // ds_load_b128
    return __builtin_shufflevector(lo, hi, 0,1,2,3,4,5,6,7,8,9,10,11,12,13,14,15);
}

// MODE 0: C[bz*cBatch + row*1024 + col] = acc                       (plain / batched)
// MODE 1: xin epilogue: row = e*64+s -> C[((e*8+bz)*64+s)*1024+col] (expert-major)
// MODE 3: y epilogue:  acc += b2[bz][col] + A[row][col]; row=b*64+s
//                      -> C[(b*1024 + bz*64 + s)*1024 + col]        (token-major)
// ABF16: A operand is bf16 in global memory (producer wrote bf16); else f32.
template<int MODE, bool TRANSA, bool ABF16>
__global__ __launch_bounds__(256)
void gemm_wmma(const void* __restrict__ Av, const float* __restrict__ Bm,
               const float* __restrict__ bias, float* __restrict__ C,
               int K, long long aBatch, long long bBatch, long long cBatch)
{
    __shared__ __align__(16) __bf16 sA[2][128 * LSTR];
    __shared__ __align__(16) __bf16 sB[2][128 * LSTR];

    const int t    = threadIdx.x;
    const int lane = t & 31;
    const int wid  = t >> 5;
    const int wr   = wid >> 2;          // 0..1  -> 64 rows each
    const int wc   = wid & 3;           // 0..3  -> 32 cols each
    const int bz   = blockIdx.z;
    const int row0 = blockIdx.y * 128;
    const int col0 = blockIdx.x * 128;

    const float*  Ab  = (const float*)Av  + (size_t)bz * aBatch;   // used if !ABF16
    const __bf16* Abh = (const __bf16*)Av + (size_t)bz * aBatch;   // used if  ABF16
    const float*  Bb  = Bm + (size_t)bz * bBatch;

    // load-thread mappings
    const int rA  = t >> 3, kvA = t & 7;   // non-transposed A: row rA+32i, k-chunk kvA
    const int kkT = t >> 5, nvT = t & 31;  // K-major reads (trans-A and B)

    float4 ra[4], rb[4];
    bf16x4 rha[4];

    auto globalLoad = [&](int k0) {
        if (ABF16) {
            if (!TRANSA) {
                #pragma unroll
                for (int i = 0; i < 4; ++i)
                    rha[i] = *reinterpret_cast<const bf16x4*>(
                        Abh + ((size_t)(row0 + rA + 32 * i) << 10) + (k0 + kvA * 4));
            } else {
                #pragma unroll
                for (int i = 0; i < 4; ++i)
                    rha[i] = *reinterpret_cast<const bf16x4*>(
                        Abh + ((size_t)(k0 + kkT + 8 * i) << 10) + (row0 + nvT * 4));
            }
        } else {
            if (!TRANSA) {
                #pragma unroll
                for (int i = 0; i < 4; ++i)
                    ra[i] = *reinterpret_cast<const float4*>(
                        Ab + ((size_t)(row0 + rA + 32 * i) << 10) + (k0 + kvA * 4));
            } else {
                #pragma unroll
                for (int i = 0; i < 4; ++i)
                    ra[i] = *reinterpret_cast<const float4*>(
                        Ab + ((size_t)(k0 + kkT + 8 * i) << 10) + (row0 + nvT * 4));
            }
        }
        #pragma unroll
        for (int i = 0; i < 4; ++i)
            rb[i] = *reinterpret_cast<const float4*>(
                Bb + ((size_t)(k0 + kkT + 8 * i) << 10) + (col0 + nvT * 4));
    };

    auto ldsStore = [&](int buf) {
        __bf16* pA = &sA[buf][0];
        __bf16* pB = &sB[buf][0];
        if (ABF16) {
            if (!TRANSA) {
                #pragma unroll
                for (int i = 0; i < 4; ++i)
                    *reinterpret_cast<bf16x4*>(&pA[(rA + 32 * i) * LSTR + kvA * 4]) = rha[i];
            } else {
                #pragma unroll
                for (int i = 0; i < 4; ++i) {
                    const int k = kkT + 8 * i;
                    pA[(nvT * 4 + 0) * LSTR + k] = rha[i][0];
                    pA[(nvT * 4 + 1) * LSTR + k] = rha[i][1];
                    pA[(nvT * 4 + 2) * LSTR + k] = rha[i][2];
                    pA[(nvT * 4 + 3) * LSTR + k] = rha[i][3];
                }
            }
        } else {
            if (!TRANSA) {
                #pragma unroll
                for (int i = 0; i < 4; ++i) {
                    // native v_cvt_pk_bf16_f32 pairs + one 8-byte ds store
                    bf16x4 q = { (__bf16)ra[i].x, (__bf16)ra[i].y,
                                 (__bf16)ra[i].z, (__bf16)ra[i].w };
                    *reinterpret_cast<bf16x4*>(&pA[(rA + 32 * i) * LSTR + kvA * 4]) = q;
                }
            } else {
                #pragma unroll
                for (int i = 0; i < 4; ++i) {
                    const int k = kkT + 8 * i;
                    pA[(nvT * 4 + 0) * LSTR + k] = (__bf16)ra[i].x;
                    pA[(nvT * 4 + 1) * LSTR + k] = (__bf16)ra[i].y;
                    pA[(nvT * 4 + 2) * LSTR + k] = (__bf16)ra[i].z;
                    pA[(nvT * 4 + 3) * LSTR + k] = (__bf16)ra[i].w;
                }
            }
        }
        #pragma unroll
        for (int i = 0; i < 4; ++i) {
            const int k = kkT + 8 * i;
            pB[(nvT * 4 + 0) * LSTR + k] = (__bf16)rb[i].x;
            pB[(nvT * 4 + 1) * LSTR + k] = (__bf16)rb[i].y;
            pB[(nvT * 4 + 2) * LSTR + k] = (__bf16)rb[i].z;
            pB[(nvT * 4 + 3) * LSTR + k] = (__bf16)rb[i].w;
        }
    };

    f32x8 acc[4][2];
    #pragma unroll
    for (int m = 0; m < 4; ++m)
        #pragma unroll
        for (int n = 0; n < 2; ++n)
            #pragma unroll
            for (int i = 0; i < 8; ++i) acc[m][n][i] = 0.0f;

    const int nt = K >> 5;
    globalLoad(0);
    ldsStore(0);
    int cur = 0;

    for (int kt = 0; kt < nt; ++kt) {
        __syncthreads();                       // buf[cur] ready; buf[cur^1] free
        const bool more = (kt + 1 < nt);
        if (more) globalLoad((kt + 1) << 5);   // issue next tile's loads early

        bf16x16 aF[4], bF[2];
        #pragma unroll
        for (int m = 0; m < 4; ++m) aF[m] = load_frag(&sA[cur][0], wr * 64 + m * 16, lane);
        #pragma unroll
        for (int n = 0; n < 2; ++n) bF[n] = load_frag(&sB[cur][0], wc * 32 + n * 16, lane);

        #pragma unroll
        for (int m = 0; m < 4; ++m)
            #pragma unroll
            for (int n = 0; n < 2; ++n)
                acc[m][n] = __builtin_amdgcn_wmma_f32_16x16x32_bf16(
                    false, aF[m], false, bF[n], (short)0, acc[m][n], false, false);

        if (more) ldsStore(cur ^ 1);
        cur ^= 1;
    }

    // ---- epilogue: per (m,n) one base address, 8 immediate-offset stores ----
    #pragma unroll
    for (int m = 0; m < 4; ++m)
        #pragma unroll
        for (int n = 0; n < 2; ++n) {
            const int rowB = row0 + wr * 64 + m * 16 + ((lane >> 4) << 3); // 8-aligned
            const int col  = col0 + wc * 32 + n * 16 + (lane & 15);
            if (MODE == 0) {
                float* p = C + (size_t)bz * cBatch + ((size_t)rowB << 10) + col;
                #pragma unroll
                for (int i = 0; i < 8; ++i) p[i << 10] = acc[m][n][i];
            } else if (MODE == 1) {            // rowB = e*64+s, bz = batch b
                const int e = rowB >> 6, s = rowB & 63;
                float* p = C + ((((size_t)e * 8 + bz) * 64 + s) << 10) + col;
                #pragma unroll
                for (int i = 0; i < 8; ++i) p[i << 10] = acc[m][n][i];
            } else {                           // MODE 3: rowB = b*64+s, bz = expert e
                const int b = rowB >> 6, s = rowB & 63;
                const float bv = bias[((size_t)bz << 10) + col];
                const float* pr = Ab + ((size_t)rowB << 10) + col;   // f32 A only
                float* p = C + (((size_t)b * 1024 + (size_t)bz * 64 + s) << 10) + col;
                #pragma unroll
                for (int i = 0; i < 8; ++i) p[i << 10] = acc[m][n][i] + bv + pr[i << 10];
            }
        }
}

// combine = softmax over z=e*s (contiguous, 1024 per (b,n)); one block per (b,n).
// Output written as bf16 (GEMM-8 A operand).
__global__ __launch_bounds__(256)
void combine_softmax(const float* __restrict__ w, __bf16* __restrict__ out) {
    __shared__ float red[256];
    const int tid = threadIdx.x;
    const size_t base = (size_t)blockIdx.x * 1024;
    float v[4];
    float mx = -3.0e38f;
    #pragma unroll
    for (int i = 0; i < 4; ++i) { v[i] = w[base + tid + i * 256]; mx = fmaxf(mx, v[i]); }
    red[tid] = mx; __syncthreads();
    for (int s2 = 128; s2 > 0; s2 >>= 1) {
        if (tid < s2) red[tid] = fmaxf(red[tid], red[tid + s2]);
        __syncthreads();
    }
    mx = red[0]; __syncthreads();
    float sum = 0.f;
    #pragma unroll
    for (int i = 0; i < 4; ++i) { v[i] = __expf(v[i] - mx); sum += v[i]; }
    red[tid] = sum; __syncthreads();
    for (int s2 = 128; s2 > 0; s2 >>= 1) {
        if (tid < s2) red[tid] += red[tid + s2];
        __syncthreads();
    }
    const float inv = 1.0f / red[0];
    #pragma unroll
    for (int i = 0; i < 4; ++i) out[base + tid + i * 256] = (__bf16)(v[i] * inv);
}

// dispatch = softmax over n (2048) of weights[b,n,es], bf16 output [b][n][es]
// (GEMM-4 transposed A operand). Each thread owns one es column.
__global__ __launch_bounds__(256)
void dispatch_softmax(const float* __restrict__ w, __bf16* __restrict__ dsp) {
    const int tid = threadIdx.x;
    const int b = blockIdx.x >> 2;
    const int es = ((blockIdx.x & 3) << 8) + tid;
    const float* p = w   + (size_t)b * (2048u * 1024u) + es;
    __bf16*      q = dsp + (size_t)b * (2048u * 1024u) + es;
    float mx = -3.0e38f, sum = 0.f;
    for (int n = 0; n < 2048; ++n) {
        const float v = p[(size_t)n << 10];
        if (v > mx) { sum *= __expf(mx - v); mx = v; }
        sum += __expf(v - mx);
    }
    const float inv = 1.0f / sum;
    for (int n = 0; n < 2048; ++n) {
        const float v = p[(size_t)n << 10];
        q[(size_t)n << 10] = (__bf16)(__expf(v - mx) * inv);
    }
}

// x2 = xin + relu( LN(h + b1) * g1 + be1 ); one block per row of 1024 (expert-major)
__global__ __launch_bounds__(256)
void ln_relu_res(const float* __restrict__ h, const float* __restrict__ xin,
                 const float* __restrict__ b1, const float* __restrict__ g1,
                 const float* __restrict__ be1, float* __restrict__ x2) {
    __shared__ float red[256];
    const int tid = threadIdx.x;
    const int r = blockIdx.x;          // [E][B*S] rows, 8192 total
    const int e = r >> 9;              // r / 512
    const size_t rb = (size_t)r << 10;
    const size_t eb = (size_t)e << 10;
    float hv[4];
    float s = 0.f;
    #pragma unroll
    for (int i = 0; i < 4; ++i) {
        const int c = tid + i * 256;
        hv[i] = h[rb + c] + b1[eb + c];
        s += hv[i];
    }
    red[tid] = s; __syncthreads();
    for (int s2 = 128; s2 > 0; s2 >>= 1) { if (tid < s2) red[tid] += red[tid + s2]; __syncthreads(); }
    const float mu = red[0] * (1.0f / 1024.0f); __syncthreads();
    float vs = 0.f;
    #pragma unroll
    for (int i = 0; i < 4; ++i) { const float d = hv[i] - mu; vs += d * d; }
    red[tid] = vs; __syncthreads();
    for (int s2 = 128; s2 > 0; s2 >>= 1) { if (tid < s2) red[tid] += red[tid + s2]; __syncthreads(); }
    const float rstd = rsqrtf(red[0] * (1.0f / 1024.0f) + 1e-5f);
    #pragma unroll
    for (int i = 0; i < 4; ++i) {
        const int c = tid + i * 256;
        const float t2 = (hv[i] - mu) * rstd * g1[eb + c] + be1[eb + c];
        x2[rb + c] = xin[rb + c] + fmaxf(t2, 0.0f);
    }
}

extern "C" void kernel_launch(void* const* d_in, const int* in_sizes, int n_in,
                              void* d_out, int out_size, void* d_ws, size_t ws_size,
                              hipStream_t stream) {
    const float* x   = (const float*)d_in[0];   // [8,2048,1024]
    const float* phi = (const float*)d_in[1];   // [1024,16,64] == [1024,1024]
    const float* W1  = (const float*)d_in[2];   // [16,1024,1024]
    const float* b1  = (const float*)d_in[3];   // [16,1024]
    const float* g1  = (const float*)d_in[4];
    const float* be1 = (const float*)d_in[5];
    const float* W2  = (const float*)d_in[6];
    const float* b2  = (const float*)d_in[7];
    float* out = (float*)d_out;                 // [8,2048,1024] f32

    char* ws = (char*)d_ws;
    const size_t MB = 1024ull * 1024ull;
    float*  weights = (float*) (ws + 0   * MB); // 64MB logits (read-only after GEMM 1)
    __bf16* combine = (__bf16*)(ws + 64  * MB); // 32MB bf16 [B][N][Z]
    float*  xin     = (float*) (ws + 96  * MB); // 32MB [E][B][S][D]
    float*  hbuf    = (float*) (ws + 128 * MB); // 32MB [E][B*S][D]
    float*  x2      = (float*) (ws + 160 * MB); // 32MB [E][B*S][D]
    __bf16* dsp     = (__bf16*)(ws + 192 * MB); // 32MB bf16 [B][N][ES]
    float*  eo      = (float*) (ws + 224 * MB); // 32MB [B][E*S][D]
    (void)in_sizes; (void)n_in; (void)out_size; (void)ws_size;

    const long long ND = 2048ll * 1024ll;       // per-batch token matrix (elements)
    const long long DD = 1024ll * 1024ll;       // per-expert weight matrix
    const long long SD = 512ll * 1024ll;        // per-expert activation matrix

    // 1) logits = x @ phi : [16384,1024] x [1024,1024]
    gemm_wmma<0, false, false><<<dim3(8, 128, 1), 256, 0, stream>>>(
        x, phi, nullptr, weights, 1024, 0, 0, 0);

    // 2) combine = softmax over e*s -> bf16
    combine_softmax<<<dim3(16384), 256, 0, stream>>>(weights, combine);

    // 3) dispatch = softmax over n -> bf16 [b][n][es]
    dispatch_softmax<<<dim3(32), 256, 0, stream>>>(weights, dsp);

    // 4) xin = dispatch^T @ x per batch b -> expert-major [E][B][S][D]
    gemm_wmma<1, true, true><<<dim3(8, 8, 8), 256, 0, stream>>>(
        dsp, x, nullptr, xin, 2048, ND, ND, 0);

    // 5) h = xin @ W1 per expert e
    gemm_wmma<0, false, false><<<dim3(8, 4, 16), 256, 0, stream>>>(
        xin, W1, nullptr, hbuf, 1024, SD, DD, SD);

    // 6) x2 = xin + relu(LN(h + b1)*g1 + be1)
    ln_relu_res<<<dim3(8192), 256, 0, stream>>>(hbuf, xin, b1, g1, be1, x2);

    // 7) expert_out = x2 + x2 @ W2 + b2, remapped to token-major [B][E*S][D]
    gemm_wmma<3, false, false><<<dim3(8, 4, 16), 256, 0, stream>>>(
        x2, W2, b2, eo, 1024, SD, DD, 0);

    // 8) out = combine @ expert_out per batch b
    gemm_wmma<0, false, true><<<dim3(8, 16, 8), 256, 0, stream>>>(
        combine, eo, nullptr, out, 1024, ND, DD, ND);
}
